// MultiHeadAttention_1735166787724
// MI455X (gfx1250) — matrix operand order repaired and checked by reading, not compiled, not run on hardware
//
#include <hip/hip_runtime.h>

// ---------------------------------------------------------------------------
// Multi-head causal attention forward for MI455X (gfx1250, wave32, WMMA).
//   B=4, T=2048, C=1024, H=16, hd=64, HEAD_SIZE=1024
// GEMMs: v_wmma_f32_16x16x32_f16.  Attention K/V staging: TDM
// tensor_load_to_lds, double-buffered so the DMA of block kb+1 overlaps the
// WMMA compute on block kb; LDS row padding produced by the D# pad fields.
// ---------------------------------------------------------------------------

typedef __attribute__((ext_vector_type(16))) _Float16     v16h;
typedef __attribute__((ext_vector_type(8)))  float        v8f;
typedef __attribute__((ext_vector_type(4)))  unsigned int v4u;
typedef __attribute__((ext_vector_type(8)))  int          v8i;
typedef __attribute__((ext_vector_type(4)))  int          v4i;

#define B_   4
#define T_   2048
#define C_   1024
#define NH_  16
#define HD_  64
#define HS_  1024            // HEAD_SIZE
#define BT_  (B_ * T_)       // 8192

// Union for assembling a v16h from two 16-byte chunks.
union F16x16 { v16h v; float4 q[2]; };

// A-fragment (16Mx32K f16): per lane (m = lane%16, half = lane/16) the 16
// elements are two contiguous 8-f16 runs at K = 8*half and K = 16 + 8*half.
__device__ __forceinline__ v16h load_afrag(const _Float16* row, int half) {
    F16x16 u;
    u.q[0] = *(const float4*)(row + 8 * half);
    u.q[1] = *(const float4*)(row + 16 + 8 * half);
    return u.v;
}
// B-fragment (32Kx16N f16): per lane (n = lane%16, half = lane/16) the 16
// elements are two contiguous 8-f16 runs at K = 8*half and K = 16 + 8*half (row given in K-major).
__device__ __forceinline__ v16h load_bfrag(const _Float16* row, int half) {
    F16x16 u;
    u.q[0] = *(const float4*)(row + 8 * half);
    u.q[1] = *(const float4*)(row + 16 + 8 * half);
    return u.v;
}

// Convert float4 -> 4xf16 and store as one 8-byte LDS write.
__device__ __forceinline__ void cvt4_store(_Float16* dst, float4 s) {
    union { _Float16 h[4]; float2 f2; } u;
    u.h[0] = (_Float16)s.x; u.h[1] = (_Float16)s.y;
    u.h[2] = (_Float16)s.z; u.h[3] = (_Float16)s.w;
    *(float2*)dst = u.f2;
}

__device__ __forceinline__ v8f wmma_f16(v16h a, v16h b, v8f c) {
    return __builtin_amdgcn_wmma_f32_16x16x32_f16(
        false, a, false, b, (short)0, c, false, false);
}

// Issue a TDM 2D tile load (f16 elements) global -> LDS.
//   tensor: [tensor_d1 rows x tensor_d0 cols], row stride = stride_d0 elems.
//   tile:   [tile_d1 rows x tile_d0 cols] starting at gaddr.
//   LDS padding: pad_amount (code) DWORDs every pad_interval (code) DWORDs.
__device__ __forceinline__ void tdm_load_2d(
    unsigned lds_addr, const void* gaddr,
    unsigned tensor_d0, unsigned tensor_d1,
    unsigned tile_d0, unsigned tile_d1, unsigned stride_d0,
    unsigned pad_interval_code, unsigned pad_amount_code)
{
    const unsigned long long ga = (unsigned long long)gaddr;
    v4u g0;
    g0[0] = 1u;                                     // count=1 valid descriptor
    g0[1] = lds_addr;                               // LDS byte address
    g0[2] = (unsigned)(ga & 0xffffffffu);           // global_addr[31:0]
    g0[3] = (unsigned)((ga >> 32) & 0x1ffffffu)     // global_addr[56:32]
          | (2u << 30);                             // type = 2 ("image")
    v8i g1;
    g1[0] = (int)((1u << 16)                        // data_size = 2 bytes
          | (1u << 20)                              // pad_enable
          | (pad_interval_code << 22)
          | (pad_amount_code << 25));
    g1[1] = (int)((tensor_d0 & 0xffffu) << 16);     // tensor_dim0 lo16
    g1[2] = (int)(((tensor_d0 >> 16) & 0xffffu)     // tensor_dim0 hi16
          | ((tensor_d1 & 0xffffu) << 16));         // tensor_dim1 lo16
    g1[3] = (int)(((tensor_d1 >> 16) & 0xffffu)     // tensor_dim1 hi16
          | (tile_d0 << 16));                       // tile_dim0
    g1[4] = (int)(tile_d1 & 0xffffu);               // tile_dim1 (tile_dim2=0)
    g1[5] = (int)stride_d0;                         // tensor_dim0_stride lo32
    g1[6] = 0;                                      // stride hi / dim1_stride
    g1[7] = 0;
    const v4i z4 = {0, 0, 0, 0};
#if __clang_major__ >= 23
    const v8i z8 = {0, 0, 0, 0, 0, 0, 0, 0};
    __builtin_amdgcn_tensor_load_to_lds(g0, g1, z4, z4, z8, 0);
#else
    __builtin_amdgcn_tensor_load_to_lds(g0, g1, z4, z4, 0);
#endif
}

// ===========================================================================
// Kernel 1: fused QKV projection.  Y = x @ W{q,k,v}  (gridDim.z selects W).
// Block = 128 threads (4 waves); tile 64(M) x 64(N); k-step 32.
// ===========================================================================
__global__ __launch_bounds__(128)
void qkv_kernel(const float* __restrict__ x,
                const float* __restrict__ Wq,
                const float* __restrict__ Wk,
                const float* __restrict__ Wv,
                _Float16* __restrict__ Qo,
                _Float16* __restrict__ Ko,
                _Float16* __restrict__ Vt)
{
    __shared__ _Float16 As[64][40];   // [m][k], 80B rows (16B aligned)
    __shared__ _Float16 Bs[64][40];   // [n][k] (transposed in LDS)

    const int t    = threadIdx.x;
    const int lane = t & 31, wave = t >> 5;
    const int half = lane >> 4, l16 = lane & 15;

    const int row0  = blockIdx.x * 64;            // over B*T
    const int n0    = blockIdx.y * 64;            // over HEAD_SIZE
    const int which = blockIdx.z;                 // 0=Q 1=K 2=V
    const float* W  = (which == 0) ? Wq : (which == 1) ? Wk : Wv;

    v8f acc[4] = {};

    for (int kc = 0; kc < C_; kc += 32) {
        float4 ra[4];
        #pragma unroll
        for (int i = 0; i < 4; ++i) {              // x tile 64x32 f32
            int idx = t + i * 128;
            int r = idx >> 3, c4 = idx & 7;
            ra[i] = *(const float4*)&x[(size_t)(row0 + r) * C_ + kc + c4 * 4];
        }
        float4 rb[4];
        #pragma unroll
        for (int i = 0; i < 4; ++i) {              // W tile 32x64 f32
            int idx = t + i * 128;
            int k = idx >> 4, n4 = idx & 15;
            rb[i] = *(const float4*)&W[(size_t)(kc + k) * HS_ + n0 + n4 * 4];
        }
        if (kc + 32 < C_) {                        // prefetch next x tile
            __builtin_prefetch(
                (const char*)&x[(size_t)(row0 + (t >> 1)) * C_ + kc + 32], 0, 0);
        }
        #pragma unroll
        for (int i = 0; i < 4; ++i) {
            int idx = t + i * 128;
            int r = idx >> 3, c4 = idx & 7;
            cvt4_store(&As[r][c4 * 4], ra[i]);
        }
        #pragma unroll
        for (int i = 0; i < 4; ++i) {              // transposed scatter
            int idx = t + i * 128;
            int k = idx >> 4, n4 = idx & 15;
            Bs[n4 * 4 + 0][k] = (_Float16)rb[i].x;
            Bs[n4 * 4 + 1][k] = (_Float16)rb[i].y;
            Bs[n4 * 4 + 2][k] = (_Float16)rb[i].z;
            Bs[n4 * 4 + 3][k] = (_Float16)rb[i].w;
        }
        __syncthreads();

        const v16h a = load_afrag(&As[wave * 16 + l16][0], half);
        #pragma unroll
        for (int nt = 0; nt < 4; ++nt) {
            const v16h b = load_bfrag(&Bs[nt * 16 + l16][0], half);
            acc[nt] = wmma_f16(a, b, acc[nt]);
        }
        __syncthreads();
    }

    // Epilogue (select hoisted out of the store loops; branch is uniform).
    const int rbase = row0 + wave * 16;
    if (which == 2) {
        #pragma unroll
        for (int nt = 0; nt < 4; ++nt) {
            const int col = n0 + nt * 16 + l16;
            const int h = col >> 6, d = col & 63;
            #pragma unroll
            for (int v = 0; v < 8; ++v) {
                const int row = rbase + v + 8 * half;
                const int bb = row >> 11, tt = row & (T_ - 1);
                Vt[((size_t)(bb * NH_ + h) * HD_ + d) * T_ + tt] =
                    (_Float16)acc[nt][v];
            }
        }
    } else {
        _Float16* __restrict__ dst = (which == 1) ? Ko : Qo;
        #pragma unroll
        for (int nt = 0; nt < 4; ++nt) {
            const int col = n0 + nt * 16 + l16;
            const int h = col >> 6, d = col & 63;
            #pragma unroll
            for (int v = 0; v < 8; ++v) {
                const int row = rbase + v + 8 * half;
                const int bb = row >> 11, tt = row & (T_ - 1);
                dst[((size_t)(bb * NH_ + h) * T_ + tt) * HD_ + d] =
                    (_Float16)acc[nt][v];
            }
        }
    }
}

// ===========================================================================
// Kernel 2: causal flash attention, one (b,h) x 64-query block per CTA.
// S^T = K·Q^T (lane owns a query column); O^T += V^T·P.
// K/V blocks double-buffered, staged by TDM tensor_load_to_lds (DMA of
// block kb+1 overlaps WMMA on block kb; one s_wait_tensorcnt + one barrier
// per iteration).
// ===========================================================================
__global__ __launch_bounds__(128)
void attn_kernel(const _Float16* __restrict__ Qg,   // [B,H,T,hd]
                 const _Float16* __restrict__ Kg,   // [B,H,T,hd]
                 const _Float16* __restrict__ Vt,   // [B,H,hd,T]
                 _Float16* __restrict__ Og)         // [B,T,HS]
{
    __shared__ _Float16 Ks[2][32][72];   // k-pos x d, 144B rows
    __shared__ _Float16 Vs[2][64][40];   // d x k-pos (V^T), 80B rows
    __shared__ _Float16 Ps[4][16][40];   // per-wave P: q x k

    const int t    = threadIdx.x;
    const int lane = t & 31, wave = t >> 5;
    const int half = lane >> 4, l16 = lane & 15;

    const int bh  = blockIdx.y;
    const int qb0 = blockIdx.x * 64;
    const int q0  = qb0 + wave * 16;

    const _Float16* Qbh = Qg + (size_t)bh * T_ * HD_;
    const _Float16* Kbh = Kg + (size_t)bh * T_ * HD_;
    const _Float16* Vbh = Vt + (size_t)bh * HD_ * T_;

    const int nkb = (qb0 >> 5) + 2;      // causal #key-blocks of 32

    // TDM staging of block 0 (wave 0 issues; D# pad makes padded LDS rows).
    //   K tile 32x64 f16: rows of 32 DWORDs (code 4) + 4 DWORD pad (code 3).
    //   V tile 64x32 f16: rows of 16 DWORDs (code 3) + 4 DWORD pad (code 3).
    if (wave == 0) {
        tdm_load_2d((unsigned)(size_t)&Ks[0][0][0], Kbh,
                    HD_, T_, HD_, 32, HD_, 4u, 3u);
        tdm_load_2d((unsigned)(size_t)&Vs[0][0][0], Vbh,
                    T_, HD_, 32, HD_, T_, 3u, 3u);
    }

    // Persistent Q B-fragments: per lane two 16B runs (K=8*half, 16+8*half) per hd-chunk.
    const _Float16* qrow = Qbh + (size_t)(q0 + l16) * HD_;
    const v16h bq0 = load_bfrag(qrow, half);
    const v16h bq1 = load_bfrag(qrow + 32, half);

    v8f   oacc[4] = {};
    float m_run = -1e30f, l_run = 0.f;
    const float sm_scale = 0.125f;       // 1/sqrt(hd)

    for (int kb = 0; kb < nkb; ++kb) {
        const int kbase = kb * 32;
        const int buf = kb & 1;

        if (wave == 0) __builtin_amdgcn_s_wait_tensorcnt(0);
        __syncthreads();                 // buf ready; buf^1 free

        if (kb + 1 < nkb) {              // DMA next block into buf^1
            const int nb = kbase + 32;
            if (wave == 0) {
                tdm_load_2d((unsigned)(size_t)&Ks[buf ^ 1][0][0],
                            Kbh + (size_t)nb * HD_,
                            HD_, T_, HD_, 32, HD_, 4u, 3u);
                tdm_load_2d((unsigned)(size_t)&Vs[buf ^ 1][0][0],
                            Vbh + nb,
                            T_, HD_, 32, HD_, T_, 3u, 3u);
            }
        }

        if (kbase <= q0 + 15) {          // wave has live keys in this block
            // --- scores S^T (two 16k x 16q tiles) ---
            v8f st[2];
            #pragma unroll
            for (int kt = 0; kt < 2; ++kt) {
                const _Float16* krow = &Ks[buf][kt * 16 + l16][0];
                const v16h a0 = load_afrag(krow, half);
                const v16h a1 = load_afrag(krow + 32, half);
                v8f s = {};
                s = wmma_f16(a0, bq0, s);
                s = wmma_f16(a1, bq1, s);
                st[kt] = s;
            }
            // --- scale + causal mask + online softmax (lane owns column q) ---
            const int qg = q0 + l16;
            float mx = -1e30f;
            #pragma unroll
            for (int kt = 0; kt < 2; ++kt)
                #pragma unroll
                for (int v = 0; v < 8; ++v) {
                    const int kg = kbase + kt * 16 + v + 8 * half;
                    float sv = (float)st[kt][v] * sm_scale;
                    sv = (kg > qg) ? -1e30f : sv;
                    st[kt][v] = sv;
                    mx = fmaxf(mx, sv);
                }
            mx = fmaxf(mx, __shfl_xor(mx, 16, 32));
            const float m_new = fmaxf(m_run, mx);
            const float scale = __expf(m_run - m_new);
            float psum = 0.f;
            #pragma unroll
            for (int kt = 0; kt < 2; ++kt)
                #pragma unroll
                for (int v = 0; v < 8; ++v) {
                    const float p = __expf((float)st[kt][v] - m_new);
                    psum += p;
                    Ps[wave][l16][kt * 16 + v + 8 * half] = (_Float16)p;
                }
            psum += __shfl_xor(psum, 16, 32);
            l_run = l_run * scale + psum;
            m_run = m_new;
            #pragma unroll
            for (int dt = 0; dt < 4; ++dt)
                #pragma unroll
                for (int v = 0; v < 8; ++v)
                    oacc[dt][v] *= scale;

            // P C-layout -> B-fragment via per-wave LDS (in-order per wave).
            const v16h bp = load_bfrag(&Ps[wave][l16][0], half);

            // O^T += V^T x P  (4 d-tiles)
            #pragma unroll
            for (int dt = 0; dt < 4; ++dt) {
                const v16h av = load_afrag(&Vs[buf][dt * 16 + l16][0], half);
                oacc[dt] = wmma_f16(av, bp, oacc[dt]);
            }
        }
    }

    // Normalize and store O (O^T layout: lane column q=l16, rows d=v+8*half).
    const float inv = 1.f / l_run;
    const int bb = bh >> 4, h = bh & 15;
    const int tq = q0 + l16;
    #pragma unroll
    for (int dt = 0; dt < 4; ++dt)
        #pragma unroll
        for (int v = 0; v < 8; ++v) {
            const int d = h * 64 + dt * 16 + v + 8 * half;
            Og[(size_t)(bb * T_ + tq) * HS_ + d] = (_Float16)(oacc[dt][v] * inv);
        }
}

// ===========================================================================
// Kernel 3: output projection  out = O @ Wc + bc   (f32 out).
// ===========================================================================
__global__ __launch_bounds__(128)
void proj_kernel(const _Float16* __restrict__ O,   // [BT, HS] f16
                 const float* __restrict__ Wc,     // [HS, C]
                 const float* __restrict__ bc,     // [C]
                 float* __restrict__ out)          // [BT, C]
{
    __shared__ _Float16 As[64][40];
    __shared__ _Float16 Bs[64][40];   // transposed [n][k]

    const int t    = threadIdx.x;
    const int lane = t & 31, wave = t >> 5;
    const int half = lane >> 4, l16 = lane & 15;

    const int row0 = blockIdx.x * 64;
    const int n0   = blockIdx.y * 64;

    v8f acc[4] = {};

    for (int kc = 0; kc < HS_; kc += 32) {
        float4 ra[2];
        #pragma unroll
        for (int i = 0; i < 2; ++i) {    // O tile 64x32 f16 (b128 copies)
            int idx = t + i * 128;
            int r = idx >> 2, c8 = idx & 3;
            ra[i] = *(const float4*)&O[(size_t)(row0 + r) * HS_ + kc + c8 * 8];
        }
        float4 rb[4];
        #pragma unroll
        for (int i = 0; i < 4; ++i) {    // Wc tile 32x64 f32
            int idx = t + i * 128;
            int k = idx >> 4, n4 = idx & 15;
            rb[i] = *(const float4*)&Wc[(size_t)(kc + k) * C_ + n0 + n4 * 4];
        }
        #pragma unroll
        for (int i = 0; i < 2; ++i) {
            int idx = t + i * 128;
            int r = idx >> 2, c8 = idx & 3;
            *(float4*)&As[r][c8 * 8] = ra[i];
        }
        #pragma unroll
        for (int i = 0; i < 4; ++i) {
            int idx = t + i * 128;
            int k = idx >> 4, n4 = idx & 15;
            Bs[n4 * 4 + 0][k] = (_Float16)rb[i].x;
            Bs[n4 * 4 + 1][k] = (_Float16)rb[i].y;
            Bs[n4 * 4 + 2][k] = (_Float16)rb[i].z;
            Bs[n4 * 4 + 3][k] = (_Float16)rb[i].w;
        }
        __syncthreads();

        const v16h a = load_afrag(&As[wave * 16 + l16][0], half);
        #pragma unroll
        for (int nt = 0; nt < 4; ++nt) {
            const v16h b = load_bfrag(&Bs[nt * 16 + l16][0], half);
            acc[nt] = wmma_f16(a, b, acc[nt]);
        }
        __syncthreads();
    }

    const int rbase = row0 + wave * 16;
    #pragma unroll
    for (int nt = 0; nt < 4; ++nt) {
        const int col = n0 + nt * 16 + l16;
        const float bias = bc[col];
        #pragma unroll
        for (int v = 0; v < 8; ++v) {
            const int row = rbase + v + 8 * half;
            out[(size_t)row * C_ + col] = acc[nt][v] + bias;
        }
    }
}

// ===========================================================================
extern "C" void kernel_launch(void* const* d_in, const int* in_sizes, int n_in,
                              void* d_out, int out_size, void* d_ws, size_t ws_size,
                              hipStream_t stream)
{
    const float* x  = (const float*)d_in[0];
    const float* Wq = (const float*)d_in[1];
    const float* Wk = (const float*)d_in[2];
    const float* Wv = (const float*)d_in[3];
    const float* Wc = (const float*)d_in[4];
    const float* bc = (const float*)d_in[5];
    float* outp     = (float*)d_out;

    // Workspace: Q,K (f16 [B,H,T,hd]) + V^T (f16 [B,H,hd,T]) + O (f16 [BT,HS])
    // = 4 * 16 MB = 64 MB.
    const size_t per = (size_t)B_ * NH_ * T_ * HD_;   // 8,388,608 elems
    _Float16* Qw = (_Float16*)d_ws;
    _Float16* Kw = Qw + per;
    _Float16* Vw = Kw + per;
    _Float16* Ow = Vw + per;

    dim3 g1(BT_ / 64, HS_ / 64, 3);
    qkv_kernel<<<g1, 128, 0, stream>>>(x, Wq, Wk, Wv, Qw, Kw, Vw);

    dim3 g2(T_ / 64, B_ * NH_);
    attn_kernel<<<g2, 128, 0, stream>>>(Qw, Kw, Vw, Ow);

    dim3 g3(BT_ / 64, C_ / 64);
    proj_kernel<<<g3, 128, 0, stream>>>(Ow, Wc, bc, outp);
}
